// GCN_63651415326972
// MI455X (gfx1250) — matrix operand order repaired and checked by reading, not compiled, not run on hardware
//
#include <hip/hip_runtime.h>
#include <hip/hip_bf16.h>

// ---------------------------------------------------------------------------
// GCN forward on MI455X (gfx1250, wave32).
// Dense projections use V_WMMA_F32_16X16X32_BF16 with a split-bf16 (hi+lo)
// 3-WMMA scheme for ~fp32 accuracy. Edge aggregation = L2-resident
// global_atomic_add_f32 scatter (feature matrices are 12.8MB << 192MB L2).
// ---------------------------------------------------------------------------

typedef __attribute__((ext_vector_type(16))) __bf16 v16bf;
typedef __attribute__((ext_vector_type(8)))  float  v8f;

#define N_IN  32
#define N_HID 64
#define N_CAT 96
#define N_OUT 64

__device__ __forceinline__ v8f wmma_bf16(v16bf a, v16bf b, v8f c) {
  // 8 args: (neg_a, A, neg_b, B, c_mod, C, reuse_a, reuse_b)
  return __builtin_amdgcn_wmma_f32_16x16x32_bf16(false, a, false, b,
                                                 (short)0, c, false, false);
}

__device__ __forceinline__ void splitbf(float x, __bf16& hi, __bf16& lo) {
  hi = (__bf16)x;
  lo = (__bf16)(x - (float)hi);
}

// ---------------------------------------------------------------- degrees ---
__global__ void k_degree(const int* __restrict__ src, const int* __restrict__ dst,
                         float* degO, float* degI, int nE) {
  int e = blockIdx.x * blockDim.x + threadIdx.x;
  if (e < nE) {
    atomicAdd(&degO[src[e]], 1.0f);
    atomicAdd(&degI[dst[e]], 1.0f);
  }
}

__global__ void k_norm(float* degO, float* degI, int n) {
  int i = blockIdx.x * blockDim.x + threadIdx.x;
  if (i < n) {
    degO[i] = 1.0f / sqrtf(fmaxf(degO[i], 1.0f));
    degI[i] = 1.0f / sqrtf(fmaxf(degI[i], 1.0f));
  }
}

// ------------------------------------------------------------------ GEMM1 ---
// Xp[n,0:64] = (X[n,0:32] @ W1[32,64]) * src_norm[n]
// One wave per 16-row tile; 4 N-tiles of 16; K=32 in a single WMMA step.
__global__ void k_gemm1(const float* __restrict__ X, const float* __restrict__ W1,
                        const float* __restrict__ srcn, float* __restrict__ Xp,
                        int nNodes) {
  __shared__ float sW[N_IN * N_HID];
  for (int i = threadIdx.x; i < N_IN * N_HID; i += blockDim.x) sW[i] = W1[i];
  __syncthreads();

  int nTiles = (nNodes + 15) >> 4;
  int wid = blockIdx.x * (blockDim.x >> 5) + (threadIdx.x >> 5);
  if (wid >= nTiles) return;            // wave-uniform: EXEC stays all-1s

  int lane = threadIdx.x & 31;
  int row0 = wid << 4;
  int m    = row0 + (lane & 15);
  int mc   = m < nNodes ? m : nNodes - 1;
  int kb   = (lane >> 4) << 4;          // A: 0 or 16

  // Pack A (16x32 bf16, split hi/lo).  Contiguous k order:
  //  lanes 0-15: elem j -> K=kb+j ; elem 8+j -> K=kb+8+j  (M = lane&15)
  v16bf ah, al;
#pragma unroll
  for (int j = 0; j < 8; ++j) {
    float x0 = X[mc * N_IN + kb + j];
    float x1 = X[mc * N_IN + kb + 8 + j];
    __bf16 h, l;
    splitbf(x0, h, l); ah[j]     = h; al[j]     = l;
    splitbf(x1, h, l); ah[8 + j] = h; al[8 + j] = l;
  }

  v8f acc[4] = {};
  int n   = lane & 15;
  int kb2 = (lane >> 4) << 4;           // B: 0 or 16
#pragma unroll
  for (int nt = 0; nt < 4; ++nt) {
    v16bf bh, bl;
#pragma unroll
    for (int i = 0; i < 16; ++i) {
      __bf16 h, l;
      splitbf(sW[(kb2 + i) * N_HID + nt * 16 + n], h, l);
      bh[i] = h; bl[i] = l;
    }
    acc[nt] = wmma_bf16(al, bh, acc[nt]);   // lo*hi
    acc[nt] = wmma_bf16(ah, bl, acc[nt]);   // hi*lo
    acc[nt] = wmma_bf16(ah, bh, acc[nt]);   // hi*hi
  }

  // C/D layout: elem r -> (M = r + 8*(lane>=16), N = lane&15)
  int rbase = row0 + ((lane >> 4) << 3);
#pragma unroll
  for (int r = 0; r < 8; ++r) {
    int row = rbase + r;
    if (row < nNodes) {
      float s = srcn[row];
#pragma unroll
      for (int nt = 0; nt < 4; ++nt)
        Xp[row * N_HID + nt * 16 + n] = acc[nt][r] * s;
    }
  }
}

// ------------------------------------------------------------------ GEMM2 ---
// Xp[n,0:64] = (concat(H1[n,0:64], X[n,0:32]) @ W2[96,64]) * src_norm[n]
__global__ void k_gemm2(const float* __restrict__ H1, const float* __restrict__ X,
                        const float* __restrict__ W2, const float* __restrict__ srcn,
                        float* __restrict__ Xp, int nNodes) {
  __shared__ float sW[N_CAT * N_OUT];   // 24 KB
  for (int i = threadIdx.x; i < N_CAT * N_OUT; i += blockDim.x) sW[i] = W2[i];
  __syncthreads();

  int nTiles = (nNodes + 15) >> 4;
  int wid = blockIdx.x * (blockDim.x >> 5) + (threadIdx.x >> 5);
  if (wid >= nTiles) return;

  int lane = threadIdx.x & 31;
  int row0 = wid << 4;
  int m    = row0 + (lane & 15);
  int mc   = m < nNodes ? m : nNodes - 1;
  int kb   = (lane >> 4) << 4;
  int n    = lane & 15;
  int kb2  = (lane >> 4) << 4;

  v8f acc[4] = {};
#pragma unroll
  for (int ks = 0; ks < 3; ++ks) {      // K steps: H1[0:32], H1[32:64], X[0:32]
    int k0 = ks * 32;
    const float* A = (ks < 2) ? H1 : X;
    int ldA  = (ks < 2) ? N_HID : N_IN;
    int koff = (ks == 1) ? 32 : 0;

    v16bf ah, al;
#pragma unroll
    for (int j = 0; j < 8; ++j) {
      float x0 = A[mc * ldA + koff + kb + j];
      float x1 = A[mc * ldA + koff + kb + 8 + j];
      __bf16 h, l;
      splitbf(x0, h, l); ah[j]     = h; al[j]     = l;
      splitbf(x1, h, l); ah[8 + j] = h; al[8 + j] = l;
    }
#pragma unroll
    for (int nt = 0; nt < 4; ++nt) {
      v16bf bh, bl;
#pragma unroll
      for (int i = 0; i < 16; ++i) {
        __bf16 h, l;
        splitbf(sW[(k0 + kb2 + i) * N_OUT + nt * 16 + n], h, l);
        bh[i] = h; bl[i] = l;
      }
      acc[nt] = wmma_bf16(al, bh, acc[nt]);
      acc[nt] = wmma_bf16(ah, bl, acc[nt]);
      acc[nt] = wmma_bf16(ah, bh, acc[nt]);
    }
  }

  int rbase = row0 + ((lane >> 4) << 3);
#pragma unroll
  for (int r = 0; r < 8; ++r) {
    int row = rbase + r;
    if (row < nNodes) {
      float s = srcn[row];
#pragma unroll
      for (int nt = 0; nt < 4; ++nt)
        Xp[row * N_OUT + nt * 16 + n] = acc[nt][r] * s;
    }
  }
}

// ----------------------------------------------------- edge scatter (L2) ----
// 16 threads per edge, 4 feats each: H[dst] += Xp[src] * w
__global__ void k_edge(const int* __restrict__ src, const int* __restrict__ dst,
                       const float* __restrict__ ew, const float* __restrict__ Xp,
                       float* __restrict__ H, int nE) {
  int t  = blockIdx.x * blockDim.x + threadIdx.x;
  int e  = t >> 4;
  int f4 = (t & 15) << 2;
  if (e >= nE) return;
  int s = src[e], d = dst[e];
  float w = ew[e];
  float4 x = *reinterpret_cast<const float4*>(Xp + (size_t)s * 64 + f4);
  float* hp = H + (size_t)d * 64 + f4;
  atomicAdd(hp + 0, x.x * w);
  atomicAdd(hp + 1, x.y * w);
  atomicAdd(hp + 2, x.z * w);
  atomicAdd(hp + 3, x.w * w);
}

// --------------------------------------------- scale + bias + ReLU (L1) -----
__global__ void k_fin1(float* __restrict__ H1, const float* __restrict__ dstn,
                       const float* __restrict__ b1, int total) {
  int i = blockIdx.x * blockDim.x + threadIdx.x;
  if (i < total) {
    int node = i >> 6, f = i & 63;
    float v = H1[i] * dstn[node] + b1[f];
    H1[i] = v > 0.0f ? v : 0.0f;
  }
}

// --------------------- scale + bias + ReLU + row L2-normalize (one wave) ----
__global__ void k_fin2(float* __restrict__ out, const float* __restrict__ dstn,
                       const float* __restrict__ b2, int nNodes) {
  int node = blockIdx.x * (blockDim.x >> 5) + (threadIdx.x >> 5);
  if (node >= nNodes) return;
  int lane = threadIdx.x & 31;
  float dn = dstn[node];
  float v0 = out[(size_t)node * 64 + lane]      * dn + b2[lane];
  float v1 = out[(size_t)node * 64 + 32 + lane] * dn + b2[32 + lane];
  v0 = fmaxf(v0, 0.0f);
  v1 = fmaxf(v1, 0.0f);
  float ss = v0 * v0 + v1 * v1;
#pragma unroll
  for (int msk = 16; msk >= 1; msk >>= 1) ss += __shfl_xor(ss, msk, 32);
  float inv = 1.0f / fmaxf(sqrtf(ss), 1e-12f);
  out[(size_t)node * 64 + lane]      = v0 * inv;
  out[(size_t)node * 64 + 32 + lane] = v1 * inv;
}

// ---------------------------------------------------------------------------
extern "C" void kernel_launch(void* const* d_in, const int* in_sizes, int n_in,
                              void* d_out, int out_size, void* d_ws, size_t ws_size,
                              hipStream_t stream) {
  const float* X   = (const float*)d_in[0];   // [N, 32]
  const float* ew  = (const float*)d_in[1];   // [E, 1]
  const int*   src = (const int*)  d_in[2];   // [E]
  const int*   dst = (const int*)  d_in[3];   // [E]
  const float* W1  = (const float*)d_in[4];   // [32, 64]
  const float* b1  = (const float*)d_in[5];   // [64]
  const float* W2  = (const float*)d_in[6];   // [96, 64]
  const float* b2  = (const float*)d_in[7];   // [64]
  float* out = (float*)d_out;

  int nNodes = in_sizes[0] / N_IN;
  int nEdges = in_sizes[2];

  // Workspace layout (floats): degO|srcn [N], degI|dstn [N], Xp [N*64], H1 [N*64]
  float* degO = (float*)d_ws;
  float* degI = degO + nNodes;
  float* Xp   = degI + nNodes;
  float* H1   = Xp + (size_t)nNodes * 64;

  hipMemsetAsync(degO, 0, (size_t)2 * nNodes * sizeof(float), stream);
  hipMemsetAsync(H1,   0, (size_t)nNodes * 64 * sizeof(float), stream);
  hipMemsetAsync(out,  0, (size_t)nNodes * 64 * sizeof(float), stream);

  // Degrees -> norms
  k_degree<<<(nEdges + 255) / 256, 256, 0, stream>>>(src, dst, degO, degI, nEdges);
  k_norm<<<(nNodes + 255) / 256, 256, 0, stream>>>(degO, degI, nNodes);

  int nTiles = (nNodes + 15) / 16;
  int gemmBlocks = (nTiles + 7) / 8;          // 8 waves / 256-thread block

  // Layer 1
  k_gemm1<<<gemmBlocks, 256, 0, stream>>>(X, W1, degO, Xp, nNodes);
  {
    long long t = (long long)nEdges * 16;
    k_edge<<<(unsigned)((t + 255) / 256), 256, 0, stream>>>(src, dst, ew, Xp, H1, nEdges);
  }
  k_fin1<<<((nNodes * 64) + 255) / 256, 256, 0, stream>>>(H1, degI, b1, nNodes * 64);

  // Layer 2 (reads H1 + X, reuses Xp, accumulates into d_out)
  k_gemm2<<<gemmBlocks, 256, 0, stream>>>(H1, X, W2, degO, Xp, nNodes);
  {
    long long t = (long long)nEdges * 16;
    k_edge<<<(unsigned)((t + 255) / 256), 256, 0, stream>>>(src, dst, ew, Xp, out, nEdges);
  }
  k_fin2<<<(nNodes + 7) / 8, 256, 0, stream>>>(out, degI, b2, nNodes);
}
